// Encoder_41815801593942
// MI455X (gfx1250) — compile-verified, matrix operand order checked by reading, not run on hardware
//
#include <hip/hip_runtime.h>
#include <math.h>

typedef __attribute__((ext_vector_type(16))) _Float16 v16h;
typedef __attribute__((ext_vector_type(8)))  float    v8f;

union Frag { v16h v; unsigned u[8]; _Float16 h[16]; };

#define SD 128
#define KPAD 160      // padded K (multiple of 32); covers msg K=151 and node K=132
#define KSTEPS 5
#define VD 12         // 4 vectors x 3 comps
#define ROWS 32       // rows (edges/nodes) per block: 2 WMMA row-tiles sharing B frags

// ---------------------------------------------------------------------------
// Weight prep: Ws [K][128] fp32 -> WsT [128][KPAD] f16 (zero padded)
// ---------------------------------------------------------------------------
__global__ __launch_bounds__(256)
void convert_wsT(const float* __restrict__ Ws, _Float16* __restrict__ WsT, int K) {
    int idx = blockIdx.x * 256 + threadIdx.x;
    if (idx >= 128 * KPAD) return;
    int c = idx / KPAD, k = idx % KPAD;
    WsT[(size_t)c * KPAD + k] = (k < K) ? (_Float16)Ws[(size_t)k * 128 + c] : (_Float16)0.f;
}

// ---------------------------------------------------------------------------
// Node scalar embedding: s = relu(concat(a_emb[at], c_emb[ch]) @ node_W + b)
// ---------------------------------------------------------------------------
__global__ __launch_bounds__(256)
void node_init_kernel(const int* __restrict__ at, const int* __restrict__ ch,
                      const float* __restrict__ a_emb, const float* __restrict__ c_emb,
                      const float* __restrict__ nW, const float* __restrict__ nb,
                      float* __restrict__ s, int N_) {
    int idx = blockIdx.x * 256 + threadIdx.x;
    if (idx >= N_ * SD) return;
    int n = idx >> 7, c = idx & 127;
    int a = at[n], q = ch[n];
    float acc = nb[c];
    #pragma unroll
    for (int j = 0; j < 16; ++j) acc += a_emb[a * 16 + j] * nW[j * SD + c];
    #pragma unroll
    for (int j = 0; j < 8; ++j)  acc += c_emb[q * 8 + j] * nW[(16 + j) * SD + c];
    s[idx] = acc > 0.f ? acc : 0.f;
}

// ---------------------------------------------------------------------------
// Edge geometry: e_feat(8) + rbf(10) -> efr[E][18]; unit dir -> xd[E][3]; degree
// ---------------------------------------------------------------------------
__global__ __launch_bounds__(256)
void edge_init_kernel(const float* __restrict__ coords, const int* __restrict__ bo,
                      const int* __restrict__ esrc, const int* __restrict__ edst,
                      const float* __restrict__ e_emb,
                      float* __restrict__ efr, float* __restrict__ xd,
                      float* __restrict__ deg, int E_) {
    int e = blockIdx.x * 256 + threadIdx.x;
    if (e >= E_) return;
    int sI = esrc[e], dI = edst[e];
    float dx = coords[(size_t)sI * 3 + 0] - coords[(size_t)dI * 3 + 0];
    float dy = coords[(size_t)sI * 3 + 1] - coords[(size_t)dI * 3 + 1];
    float dz = coords[(size_t)sI * 3 + 2] - coords[(size_t)dI * 3 + 2];
    float dist = sqrtf(dx * dx + dy * dy + dz * dz + 1e-8f);
    float inv = 1.f / dist;
    xd[(size_t)e * 3 + 0] = dx * inv;
    xd[(size_t)e * 3 + 1] = dy * inv;
    xd[(size_t)e * 3 + 2] = dz * inv;
    int b = bo[e];
    #pragma unroll
    for (int j = 0; j < 8; ++j) efr[(size_t)e * 18 + j] = e_emb[b * 8 + j];
    const float sigma = 32.0f / 10.0f;
    #pragma unroll
    for (int m = 0; m < 10; ++m) {
        float mu = 32.0f * (float)m / 9.0f;
        float z = (dist - mu) / sigma;
        efr[(size_t)e * 18 + 8 + m] = expf(-z * z);
    }
    atomicAdd(&deg[dI], 1.0f);
}

__global__ __launch_bounds__(256)
void inv_denom_kernel(const float* __restrict__ deg, float* __restrict__ inv, int N_) {
    int n = blockIdx.x * 256 + threadIdx.x;
    if (n >= N_) return;
    float d = deg[n];
    inv[n] = 1.0f / (d > 1.0f ? d : 1.0f);
}

// ---------------------------------------------------------------------------
// Message GVP: 32 edges/block, WMMA [2x16 x 151(->160)] @ [160 x 128]
// Two row-tiles per wave share each B fragment load.
// ---------------------------------------------------------------------------
__global__ __launch_bounds__(256)
void msg_gvp_kernel(const float* __restrict__ s, const float* __restrict__ V,
                    const float* __restrict__ efr, const float* __restrict__ xd,
                    const int* __restrict__ esrc, const int* __restrict__ edst,
                    const float* __restrict__ invden,
                    const _Float16* __restrict__ WsT,
                    const float* __restrict__ Wh, const float* __restrict__ Wv,
                    const float* __restrict__ Wg, const float* __restrict__ bg,
                    const float* __restrict__ bs,
                    float* __restrict__ s_acc, float* __restrict__ v_acc, int E_) {
    __shared__ _Float16 aT[ROWS][KPAD];
    __shared__ float souT[ROWS][SD];
    __shared__ float VhS[ROWS][15];
    __shared__ float gateS[ROWS][4];
    __shared__ float invS[ROWS];
    __shared__ int   dstS[ROWS];

    const int t = threadIdx.x;
    const int e0 = blockIdx.x * ROWS;

    // A-tile: k 0..127 = s[src]
    for (int idx = t; idx < ROWS * SD; idx += 256) {
        int r = idx >> 7, c = idx & 127;
        aT[r][c] = (e0 + r < E_) ? (_Float16)s[(size_t)esrc[e0 + r] * SD + c]
                                 : (_Float16)0.f;
    }
    // A-tile: k 128..145 = e_feat ++ rbf
    for (int idx = t; idx < ROWS * 18; idx += 256) {
        int r = idx / 18, j = idx % 18;
        aT[r][SD + j] = (e0 + r < E_) ? (_Float16)efr[(size_t)(e0 + r) * 18 + j]
                                      : (_Float16)0.f;
    }
    // per-edge small tensor work: Vh = mv @ Wh, vn; k 146..150 = vn, pad 0
    if (t < ROWS) {
        int e = e0 + t;
        if (e < E_) {
            int src = esrc[e], dst = edst[e];
            dstS[t] = dst;
            invS[t] = invden[dst];
            float mv[5][3];
            #pragma unroll
            for (int v = 0; v < 4; ++v)
                #pragma unroll
                for (int i = 0; i < 3; ++i)
                    mv[v][i] = V[(size_t)src * VD + v * 3 + i];
            #pragma unroll
            for (int i = 0; i < 3; ++i) mv[4][i] = xd[(size_t)e * 3 + i];
            #pragma unroll
            for (int h = 0; h < 5; ++h) {
                float v0 = 0.f, v1 = 0.f, v2 = 0.f;
                #pragma unroll
                for (int v = 0; v < 5; ++v) {
                    float w = Wh[v * 5 + h];
                    v0 += mv[v][0] * w; v1 += mv[v][1] * w; v2 += mv[v][2] * w;
                }
                VhS[t][h * 3 + 0] = v0; VhS[t][h * 3 + 1] = v1; VhS[t][h * 3 + 2] = v2;
                aT[t][146 + h] = (_Float16)sqrtf(v0 * v0 + v1 * v1 + v2 * v2 + 1e-8f);
            }
        } else {
            dstS[t] = 0; invS[t] = 0.f;
            #pragma unroll
            for (int h = 0; h < 15; ++h) VhS[t][h] = 0.f;
            for (int h = 0; h < 5; ++h) aT[t][146 + h] = (_Float16)0.f;
        }
        for (int k = 151; k < KPAD; ++k) aT[t][k] = (_Float16)0.f;
    }
    __syncthreads();

    // WMMA: each wave owns a 16-col tile; 2 row-tiles share each B fragment
    const int lane = t & 31, wave = t >> 5;
    const int rr = lane & 15, khalf = lane >> 4;
    const int col = wave * 16 + rr;
    v8f acc0 = {0.f, 0.f, 0.f, 0.f, 0.f, 0.f, 0.f, 0.f};
    v8f acc1 = {0.f, 0.f, 0.f, 0.f, 0.f, 0.f, 0.f, 0.f};
    const unsigned* au0 = (const unsigned*)(&aT[rr][0]);
    const unsigned* au1 = (const unsigned*)(&aT[16 + rr][0]);
    const unsigned* bu  = (const unsigned*)(WsT + (size_t)col * KPAD);
    #pragma unroll
    for (int ks = 0; ks < KSTEPS; ++ks) {
        Frag a0, a1, b;
        #pragma unroll
        for (int v = 0; v < 8; ++v) {
            int uidx = ks * 16 + (v >> 2) * 8 + khalf * 4 + (v & 3);
            a0.u[v] = au0[uidx];
            a1.u[v] = au1[uidx];
            b.u[v]  = bu[uidx];
        }
        acc0 = __builtin_amdgcn_wmma_f32_16x16x32_f16(false, a0.v, false, b.v,
                                                      (short)0, acc0, false, false);
        acc1 = __builtin_amdgcn_wmma_f32_16x16x32_f16(false, a1.v, false, b.v,
                                                      (short)0, acc1, false, false);
    }
    {
        float bsv = bs[col];
        int mbase = khalf * 8;
        #pragma unroll
        for (int j = 0; j < 8; ++j) {
            float o0 = acc0[j] + bsv;
            float o1 = acc1[j] + bsv;
            souT[mbase + j][col]      = o0 > 0.f ? o0 : 0.f;
            souT[16 + mbase + j][col] = o1 > 0.f ? o1 : 0.f;
        }
    }
    __syncthreads();

    // gate = sigmoid(s_out @ Wg + bg)   (32 edges x 4 gates)
    if (t < ROWS * 4) {
        int e = t >> 2, u = t & 3;
        float g = bg[u];
        for (int c = 0; c < SD; ++c) g += souT[e][c] * Wg[c * 4 + u];
        gateS[e][u] = 1.f / (1.f + expf(-g));
    }
    __syncthreads();

    // scatter scalar messages (pre-scaled by 1/deg(dst))
    for (int idx = t; idx < ROWS * SD; idx += 256) {
        int e = idx >> 7, c = idx & 127;
        if (e0 + e < E_)
            atomicAdd(&s_acc[(size_t)dstS[e] * SD + c], souT[e][c] * invS[e]);
    }
    // scatter gated vector messages: Vu = Vh @ Wv
    for (int idx = t; idx < ROWS * VD; idx += 256) {
        int e = idx / VD, r = idx % VD, u = r / 3, i = r % 3;
        if (e0 + e < E_) {
            float vu = 0.f;
            #pragma unroll
            for (int h = 0; h < 5; ++h) vu += VhS[e][h * 3 + i] * Wv[h * 4 + u];
            atomicAdd(&v_acc[(size_t)dstS[e] * VD + r], vu * gateS[e][u] * invS[e]);
        }
    }
}

// ---------------------------------------------------------------------------
// Node GVP + residual: 32 nodes/block, WMMA [2x16 x 132(->160)] @ [160 x 128]
// Consumes and re-zeros the message accumulators.
// ---------------------------------------------------------------------------
__global__ __launch_bounds__(256)
void node_gvp_kernel(float* __restrict__ s, float* __restrict__ V,
                     float* __restrict__ s_acc, float* __restrict__ v_acc,
                     const _Float16* __restrict__ WsT,
                     const float* __restrict__ Wh, const float* __restrict__ Wv,
                     const float* __restrict__ Wg, const float* __restrict__ bg,
                     const float* __restrict__ bs, int N_) {
    __shared__ _Float16 aT[ROWS][KPAD];
    __shared__ float s1T[ROWS][SD];
    __shared__ float souT[ROWS][SD];
    __shared__ float VhS[ROWS][VD];
    __shared__ float V1S[ROWS][VD];
    __shared__ float gateS[ROWS][4];

    const int t = threadIdx.x;
    const int n0 = blockIdx.x * ROWS;

    for (int idx = t; idx < ROWS * SD; idx += 256) {
        int r = idx >> 7, c = idx & 127;
        float s1 = 0.f;
        if (n0 + r < N_) {
            size_t g = (size_t)(n0 + r) * SD + c;
            float m = s_acc[g]; s_acc[g] = 0.f;      // consume + rezero for next layer
            s1 = s[g] + m;
        }
        s1T[r][c] = s1;
        aT[r][c] = (_Float16)s1;
    }
    if (t < ROWS) {
        int n = n0 + t;
        float v1[VD];
        if (n < N_) {
            #pragma unroll
            for (int r = 0; r < VD; ++r) {
                size_t g = (size_t)n * VD + r;
                float m = v_acc[g]; v_acc[g] = 0.f;
                v1[r] = V[g] + m;
                V1S[t][r] = v1[r];
            }
        } else {
            #pragma unroll
            for (int r = 0; r < VD; ++r) { v1[r] = 0.f; V1S[t][r] = 0.f; }
        }
        #pragma unroll
        for (int h = 0; h < 4; ++h) {
            float a0 = 0.f, a1 = 0.f, a2 = 0.f;
            #pragma unroll
            for (int v = 0; v < 4; ++v) {
                float w = Wh[v * 4 + h];
                a0 += v1[v * 3 + 0] * w; a1 += v1[v * 3 + 1] * w; a2 += v1[v * 3 + 2] * w;
            }
            VhS[t][h * 3 + 0] = a0; VhS[t][h * 3 + 1] = a1; VhS[t][h * 3 + 2] = a2;
            aT[t][SD + h] = (_Float16)sqrtf(a0 * a0 + a1 * a1 + a2 * a2 + 1e-8f);
        }
        for (int k = 132; k < KPAD; ++k) aT[t][k] = (_Float16)0.f;
    }
    __syncthreads();

    const int lane = t & 31, wave = t >> 5;
    const int rr = lane & 15, khalf = lane >> 4;
    const int col = wave * 16 + rr;
    v8f acc0 = {0.f, 0.f, 0.f, 0.f, 0.f, 0.f, 0.f, 0.f};
    v8f acc1 = {0.f, 0.f, 0.f, 0.f, 0.f, 0.f, 0.f, 0.f};
    const unsigned* au0 = (const unsigned*)(&aT[rr][0]);
    const unsigned* au1 = (const unsigned*)(&aT[16 + rr][0]);
    const unsigned* bu  = (const unsigned*)(WsT + (size_t)col * KPAD);
    #pragma unroll
    for (int ks = 0; ks < KSTEPS; ++ks) {
        Frag a0, a1, b;
        #pragma unroll
        for (int v = 0; v < 8; ++v) {
            int uidx = ks * 16 + (v >> 2) * 8 + khalf * 4 + (v & 3);
            a0.u[v] = au0[uidx];
            a1.u[v] = au1[uidx];
            b.u[v]  = bu[uidx];
        }
        acc0 = __builtin_amdgcn_wmma_f32_16x16x32_f16(false, a0.v, false, b.v,
                                                      (short)0, acc0, false, false);
        acc1 = __builtin_amdgcn_wmma_f32_16x16x32_f16(false, a1.v, false, b.v,
                                                      (short)0, acc1, false, false);
    }
    {
        float bsv = bs[col];
        int mbase = khalf * 8;
        #pragma unroll
        for (int j = 0; j < 8; ++j) {
            float o0 = acc0[j] + bsv;
            float o1 = acc1[j] + bsv;
            souT[mbase + j][col]      = o0 > 0.f ? o0 : 0.f;
            souT[16 + mbase + j][col] = o1 > 0.f ? o1 : 0.f;
        }
    }
    __syncthreads();

    if (t < ROWS * 4) {
        int e = t >> 2, u = t & 3;
        float g = bg[u];
        for (int c = 0; c < SD; ++c) g += souT[e][c] * Wg[c * 4 + u];
        gateS[e][u] = 1.f / (1.f + expf(-g));
    }
    __syncthreads();

    for (int idx = t; idx < ROWS * SD; idx += 256) {
        int r = idx >> 7, c = idx & 127;
        if (n0 + r < N_)
            s[(size_t)(n0 + r) * SD + c] = s1T[r][c] + souT[r][c];
    }
    for (int idx = t; idx < ROWS * VD; idx += 256) {
        int e = idx / VD, r = idx % VD, u = r / 3, i = r % 3;
        if (n0 + e < N_) {
            float vu = 0.f;
            #pragma unroll
            for (int h = 0; h < 4; ++h) vu += VhS[e][h * 3 + i] * Wv[h * 4 + u];
            V[(size_t)(n0 + e) * VD + r] = V1S[e][r] + vu * gateS[e][u];
        }
    }
}

// ---------------------------------------------------------------------------
// Latent head: out = relu(s @ W1 + b1) @ W2 + b2    (128 -> 16 -> 8)
// ---------------------------------------------------------------------------
__global__ __launch_bounds__(256)
void latent_kernel(const float* __restrict__ s,
                   const float* __restrict__ W1, const float* __restrict__ b1,
                   const float* __restrict__ W2, const float* __restrict__ b2,
                   float* __restrict__ out, int N_) {
    __shared__ float sl[16][SD];
    __shared__ float l1[16][16];
    const int t = threadIdx.x;
    const int n0 = blockIdx.x * 16;
    for (int idx = t; idx < 16 * SD; idx += 256) {
        int n = idx >> 7, c = idx & 127;
        sl[n][c] = (n0 + n < N_) ? s[(size_t)(n0 + n) * SD + c] : 0.f;
    }
    __syncthreads();
    {
        int n = t >> 4, k = t & 15;
        float a = b1[k];
        for (int c = 0; c < SD; ++c) a += sl[n][c] * W1[c * 16 + k];
        l1[n][k] = a > 0.f ? a : 0.f;
    }
    __syncthreads();
    if (t < 128) {
        int n = t >> 3, j = t & 7;
        float a = b2[j];
        #pragma unroll
        for (int k = 0; k < 16; ++k) a += l1[n][k] * W2[k * 8 + j];
        if (n0 + n < N_) out[(size_t)(n0 + n) * 8 + j] = a;
    }
}

// ---------------------------------------------------------------------------
// Host launcher
// ---------------------------------------------------------------------------
extern "C" void kernel_launch(void* const* d_in, const int* in_sizes, int n_in,
                              void* d_out, int out_size, void* d_ws, size_t ws_size,
                              hipStream_t stream) {
    const int N = in_sizes[0];   // 100000
    const int E = in_sizes[2];   // 400000
    (void)n_in;

    const int*   at     = (const int*)d_in[0];
    const int*   ch     = (const int*)d_in[1];
    const int*   bo     = (const int*)d_in[2];
    const float* coords = (const float*)d_in[3];
    const int*   esrc   = (const int*)d_in[4];
    const int*   edst   = (const int*)d_in[5];
    // params, jax pytree flatten order (sorted dict keys):
    const float* a_emb  = (const float*)d_in[6];
    const float* c_emb  = (const float*)d_in[7];
    const float* e_emb  = (const float*)d_in[8];
    const float* lat_W1 = (const float*)d_in[9];
    const float* lat_b1 = (const float*)d_in[10];
    const float* lat_W2 = (const float*)d_in[11];
    const float* lat_b2 = (const float*)d_in[12];
    // d_in[13 + 12*l + {0..5}]  = layer l msg  {Wg, Wh, Ws, Wv, bg, bs}
    // d_in[13 + 12*l + {6..11}] = layer l node {Wg, Wh, Ws, Wv, bg, bs}
    const float* node_W = (const float*)d_in[49];
    const float* node_b = (const float*)d_in[50];

    // workspace layout (floats)
    float* ws = (float*)d_ws;
    size_t o = 0;
    float* s_   = ws + o; o += (size_t)N * SD;
    float* V_   = ws + o; o += (size_t)N * VD;
    float* sac  = ws + o; o += (size_t)N * SD;
    float* vac  = ws + o; o += (size_t)N * VD;
    float* deg  = ws + o; o += (size_t)N;
    float* inv  = ws + o; o += (size_t)N;
    float* efr  = ws + o; o += (size_t)E * 18;
    float* xd   = ws + o; o += (size_t)E * 3;
    _Float16* wsT = (_Float16*)(ws + o);
    size_t need = o * sizeof(float) + (size_t)6 * 128 * KPAD * sizeof(_Float16);
    if (ws_size < need) return;  // workspace too small; nothing safe to do

    // zero V, s_acc, v_acc, deg (contiguous region)
    hipMemsetAsync(V_, 0, (size_t)(N * VD + N * SD + N * VD + N) * sizeof(float), stream);

    // weight prep: 6 transposed f16 Ws matrices
    const int cvtBlocks = (128 * KPAD + 255) / 256;
    for (int l = 0; l < 3; ++l) {
        int m = 13 + l * 12;
        convert_wsT<<<cvtBlocks, 256, 0, stream>>>(
            (const float*)d_in[m + 2], wsT + (size_t)(2 * l) * 128 * KPAD, 151);
        convert_wsT<<<cvtBlocks, 256, 0, stream>>>(
            (const float*)d_in[m + 8], wsT + (size_t)(2 * l + 1) * 128 * KPAD, 132);
    }

    node_init_kernel<<<(N * SD + 255) / 256, 256, 0, stream>>>(
        at, ch, a_emb, c_emb, node_W, node_b, s_, N);
    edge_init_kernel<<<(E + 255) / 256, 256, 0, stream>>>(
        coords, bo, esrc, edst, e_emb, efr, xd, deg, E);
    inv_denom_kernel<<<(N + 255) / 256, 256, 0, stream>>>(deg, inv, N);

    for (int l = 0; l < 3; ++l) {
        int m = 13 + l * 12;
        msg_gvp_kernel<<<(E + ROWS - 1) / ROWS, 256, 0, stream>>>(
            s_, V_, efr, xd, esrc, edst, inv,
            wsT + (size_t)(2 * l) * 128 * KPAD,
            (const float*)d_in[m + 1],   // Wh
            (const float*)d_in[m + 3],   // Wv
            (const float*)d_in[m + 0],   // Wg
            (const float*)d_in[m + 4],   // bg
            (const float*)d_in[m + 5],   // bs
            sac, vac, E);
        node_gvp_kernel<<<(N + ROWS - 1) / ROWS, 256, 0, stream>>>(
            s_, V_, sac, vac,
            wsT + (size_t)(2 * l + 1) * 128 * KPAD,
            (const float*)d_in[m + 7],   // Wh
            (const float*)d_in[m + 9],   // Wv
            (const float*)d_in[m + 6],   // Wg
            (const float*)d_in[m + 10],  // bg
            (const float*)d_in[m + 11],  // bs
            N);
    }

    latent_kernel<<<(N + 15) / 16, 256, 0, stream>>>(
        s_, lat_W1, lat_b1, lat_W2, lat_b2, (float*)d_out, N);

    // mask output (all false / 0)
    long long rem = (long long)out_size - (long long)N * 8;
    if (rem > 0)
        hipMemsetAsync((char*)d_out + (size_t)N * 8 * sizeof(float), 0,
                       (size_t)rem * sizeof(float), stream);
}
